// Net_5738076308259
// MI455X (gfx1250) — compile-verified, matrix operand order checked by reading, not run on hardware
//
#include <hip/hip_runtime.h>
#include <math.h>

// ---------------------------------------------------------------------------
// Types for CDNA5 WMMA (wave32): bf16 A/B fragments (16 halfs), f32 C/D (8 f)
// ---------------------------------------------------------------------------
typedef __attribute__((ext_vector_type(16))) __bf16 v16bf;
typedef __attribute__((ext_vector_type(8)))  float  v8f;
typedef int gat_vi4 __attribute__((vector_size(16)));   // async-LDS operand type

union Frag16 { v16bf v; uint4 q[2]; };

#if defined(__AMDGCN__) &&                                                    \
    __has_builtin(__builtin_amdgcn_global_load_async_to_lds_b128) &&          \
    __has_builtin(__builtin_amdgcn_s_wait_asynccnt)
#define GAT_ASYNC_LDS 1
#endif

__device__ __forceinline__ void atomicMaxF(float* addr, float v) {
  // works for all non-NaN floats when *addr is initialized to -inf
  if (v >= 0.0f) atomicMax((int*)addr, __float_as_int(v));
  else           atomicMin((unsigned int*)addr, __float_as_uint(v));
}

__device__ __forceinline__ unsigned short f32_to_bf16_bits(float f) {
  unsigned int u = __float_as_uint(f);
  unsigned int r = 0x7FFFu + ((u >> 16) & 1u);   // round-to-nearest-even
  return (unsigned short)((u + r) >> 16);
}

// ---------------------------------------------------------------------------
// Elementwise utilities
// ---------------------------------------------------------------------------
__global__ void k_fill_f32(float* __restrict__ p, float v, int n) {
  int i = blockIdx.x * blockDim.x + threadIdx.x;
  if (i < n) p[i] = v;
}

__global__ void k_f32_to_bf16(const float* __restrict__ in,
                              unsigned short* __restrict__ out, int n) {
  int i = blockIdx.x * blockDim.x + threadIdx.x;
  if (i < n) out[i] = f32_to_bf16_bits(in[i]);
}

// Tiled convert+transpose: in [K,N] f32 -> out [N,K] bf16. K,N multiples of 32.
__global__ __launch_bounds__(256) void k_f32_to_bf16_t(
    const float* __restrict__ in, unsigned short* __restrict__ out,
    int K, int N) {
  __shared__ float tile[32][33];
  int nb = blockIdx.x * 32, kb = blockIdx.y * 32;
  int tx = threadIdx.x & 31, ty = threadIdx.x >> 5;   // 32 x 8
  for (int i = ty; i < 32; i += 8)
    tile[i][tx] = in[(size_t)(kb + i) * N + nb + tx];
  __syncthreads();
  for (int i = ty; i < 32; i += 8)
    out[(size_t)(nb + i) * K + kb + tx] = f32_to_bf16_bits(tile[tx][i]);
}

// out = elu(acc + b[f] + skip + lb[f])
__global__ void k_epilogue_elu(const float* __restrict__ acc,
                               const float* __restrict__ skip,
                               const float* __restrict__ b,
                               const float* __restrict__ lb,
                               float* __restrict__ out, int total, int F) {
  int i = blockIdx.x * blockDim.x + threadIdx.x;
  if (i >= total) return;
  int f = i % F;
  float v = acc[i] + b[f] + skip[i] + lb[f];
  out[i] = v > 0.0f ? v : expm1f(v);
}

// ---------------------------------------------------------------------------
// bf16 WMMA GEMM: C[M,N] = A[M,K] * Bt[N,K]^T, f32 accumulate.
// Block tile 64x128, 256 threads = 8 waves (2x4); each wave owns a 32x32
// slab = 2x2 WMMA tiles (4 v_wmma per 32-K step, BK=64 -> 8 per LDS stage).
// Ping-pong LDS with async global->LDS staging when available.
// ---------------------------------------------------------------------------
#define GEMM_BM 64
#define GEMM_BN 128
#define GEMM_BK 64
#define LDS_STR 72   // halfs; 144B rows: 16B-aligned chunks, bank-spread

__global__ __launch_bounds__(256) void k_gemm_bf16_wmma(
    const unsigned short* __restrict__ A,    // [M,K] bf16 row-major
    const unsigned short* __restrict__ Bt,   // [N,K] bf16 row-major (B^T)
    float* __restrict__ C, int M, int N, int K) {
  __shared__ __align__(16) unsigned short As[2][GEMM_BM * LDS_STR];
  __shared__ __align__(16) unsigned short Bs[2][GEMM_BN * LDS_STR];

  const int m0 = blockIdx.x * GEMM_BM;
  const int n0 = blockIdx.y * GEMM_BN;
  const int t    = threadIdx.x;
  const int wave = t >> 5;
  const int lane = t & 31;
  const int wr0  = (wave >> 2) * 32;        // 0 / 32
  const int wc0  = (wave & 3) * 32;         // 0,32,64,96
  const int mloc = lane & 15;               // M/N index within a 16x16 tile
  const int kb   = (lane >> 4) * 8;         // K sub-block base: 0 or 8

  // staging index math (per thread)
  const int ar = t >> 2,  aseg = (t & 3) * 16;   // A: row 0..63, 16-half seg
  const int br = t >> 1,  bseg = (t & 1) * 32;   // B: row 0..127, 32-half seg

  auto stage = [&](int buf, int k0) {
#ifdef GAT_ASYNC_LDS
    {
      const unsigned short* ga = A + (size_t)(m0 + ar) * K + k0 + aseg;
      unsigned short* la = &As[buf][ar * LDS_STR + aseg];
#pragma unroll
      for (int j = 0; j < 2; ++j)
        __builtin_amdgcn_global_load_async_to_lds_b128(
            (gat_vi4*)(ga + j * 8), (gat_vi4*)(la + j * 8), 0, 0);
      const unsigned short* gb = Bt + (size_t)(n0 + br) * K + k0 + bseg;
      unsigned short* lb = &Bs[buf][br * LDS_STR + bseg];
#pragma unroll
      for (int j = 0; j < 4; ++j)
        __builtin_amdgcn_global_load_async_to_lds_b128(
            (gat_vi4*)(gb + j * 8), (gat_vi4*)(lb + j * 8), 0, 0);
    }
#else
    {
      uint4 av0 = {0u, 0u, 0u, 0u}, av1 = {0u, 0u, 0u, 0u};
      int gr = m0 + ar;
      if (gr < M) {
        const uint4* gp = (const uint4*)(A + (size_t)gr * K + k0 + aseg);
        av0 = gp[0]; av1 = gp[1];
      }
      uint4* lp = (uint4*)(&As[buf][ar * LDS_STR + aseg]);
      lp[0] = av0; lp[1] = av1;
      const uint4* gb = (const uint4*)(Bt + (size_t)(n0 + br) * K + k0 + bseg);
      uint4* lq = (uint4*)(&Bs[buf][br * LDS_STR + bseg]);
#pragma unroll
      for (int j = 0; j < 4; ++j) lq[j] = gb[j];
    }
#endif
  };

  v8f acc00 = {}, acc01 = {}, acc10 = {}, acc11 = {};

  stage(0, 0);
#ifdef GAT_ASYNC_LDS
  __builtin_amdgcn_s_wait_asynccnt(0);
#endif
  __syncthreads();

  int cur = 0;
  for (int k0 = 0; k0 < K; k0 += GEMM_BK) {
    if (k0 + GEMM_BK < K) stage(cur ^ 1, k0 + GEMM_BK);

    const unsigned short* as = &As[cur][0];
    const unsigned short* bs = &Bs[cur][0];
#pragma unroll
    for (int kk = 0; kk < GEMM_BK; kk += 32) {
      Frag16 a0, a1, b0, b1;
      const int ka = kk + kb;
      a0.q[0] = *(const uint4*)(&as[(wr0 + mloc) * LDS_STR + ka]);
      a0.q[1] = *(const uint4*)(&as[(wr0 + mloc) * LDS_STR + ka + 16]);
      a1.q[0] = *(const uint4*)(&as[(wr0 + 16 + mloc) * LDS_STR + ka]);
      a1.q[1] = *(const uint4*)(&as[(wr0 + 16 + mloc) * LDS_STR + ka + 16]);
      b0.q[0] = *(const uint4*)(&bs[(wc0 + mloc) * LDS_STR + ka]);
      b0.q[1] = *(const uint4*)(&bs[(wc0 + mloc) * LDS_STR + ka + 16]);
      b1.q[0] = *(const uint4*)(&bs[(wc0 + 16 + mloc) * LDS_STR + ka]);
      b1.q[1] = *(const uint4*)(&bs[(wc0 + 16 + mloc) * LDS_STR + ka + 16]);
      acc00 = __builtin_amdgcn_wmma_f32_16x16x32_bf16(
          false, a0.v, false, b0.v, (short)0, acc00, false, false);
      acc01 = __builtin_amdgcn_wmma_f32_16x16x32_bf16(
          false, a0.v, false, b1.v, (short)0, acc01, false, false);
      acc10 = __builtin_amdgcn_wmma_f32_16x16x32_bf16(
          false, a1.v, false, b0.v, (short)0, acc10, false, false);
      acc11 = __builtin_amdgcn_wmma_f32_16x16x32_bf16(
          false, a1.v, false, b1.v, (short)0, acc11, false, false);
    }
#ifdef GAT_ASYNC_LDS
    __builtin_amdgcn_s_wait_asynccnt(0);
#endif
    __syncthreads();
    cur ^= 1;
  }

  // C/D layout: VGPR v -> tile row v + 8*(lane>>4), tile col = lane&15
  const int c0 = n0 + wc0 + mloc;
  const int c1 = c0 + 16;
  const int rb = m0 + wr0 + (lane >> 4) * 8;
#pragma unroll
  for (int v = 0; v < 8; ++v) {
    int r0 = rb + v;
    if (r0 < M) {
      C[(size_t)r0 * N + c0] = acc00[v];
      C[(size_t)r0 * N + c1] = acc01[v];
    }
    int r1 = r0 + 16;
    if (r1 < M) {
      C[(size_t)r1 * N + c0] = acc10[v];
      C[(size_t)r1 * N + c1] = acc11[v];
    }
  }
}

// ---------------------------------------------------------------------------
// Small-K GEMM (conv1: K=8). One thread per output element.
// ---------------------------------------------------------------------------
__global__ void k_gemm_smallk(const float* __restrict__ A,  // [M,K]
                              const float* __restrict__ W,  // [K,N]
                              float* __restrict__ C, int M, int N, int K) {
  int idx = blockIdx.x * blockDim.x + threadIdx.x;
  if (idx >= M * N) return;
  int n = idx / N, j = idx - n * N;
  float s = 0.0f;
  for (int k = 0; k < K; ++k) s += A[n * K + k] * W[k * N + j];
  C[idx] = s;
}

// ---------------------------------------------------------------------------
// Per-node attention halves: sn[n,h] = <xl[n,h,:], a_src[h,:]>, dn likewise.
// One wave per (node, head).
// ---------------------------------------------------------------------------
__global__ __launch_bounds__(256) void k_node_attn(
    const float* __restrict__ xl, const float* __restrict__ a_s,
    const float* __restrict__ a_d, float* __restrict__ sn,
    float* __restrict__ dn, int Nn, int H, int C) {
  int wid  = blockIdx.x * 8 + (threadIdx.x >> 5);
  int lane = threadIdx.x & 31;
  if (wid >= Nn * H) return;
  int n = wid / H, h = wid - n * H;
  const float* row = xl + (size_t)n * H * C + (size_t)h * C;
  float s = 0.0f, d = 0.0f;
  for (int c = lane; c < C; c += 32) {
    float v = row[c];
    s += v * a_s[h * C + c];
    d += v * a_d[h * C + c];
  }
  for (int off = 16; off; off >>= 1) {
    s += __shfl_down(s, off, 32);
    d += __shfl_down(d, off, 32);
  }
  if (lane == 0) { sn[wid] = s; dn[wid] = d; }
}

// ---------------------------------------------------------------------------
// Edge pass 1: e = leakyrelu(sn[src]+dn[dst]); segment max over dst (atomic).
// ---------------------------------------------------------------------------
__global__ void k_edge_score_max(const long long* __restrict__ ei, int E,
                                 int Etot, const float* __restrict__ sn,
                                 const float* __restrict__ dn,
                                 float* __restrict__ eb,
                                 float* __restrict__ mb, int H) {
  int idx = blockIdx.x * blockDim.x + threadIdx.x;
  if (idx >= Etot * H) return;
  int e = idx / H, h = idx - e * H;
  int src, dst;
  if (e < E) { src = (int)ei[e]; dst = (int)ei[E + e]; }
  else       { src = dst = e - E; }                      // self-loops
  float v = sn[src * H + h] + dn[dst * H + h];
  v = v > 0.0f ? v : 0.2f * v;                           // LeakyReLU(0.2)
  eb[idx] = v;
  atomicMaxF(&mb[dst * H + h], v);
}

// Edge pass 2: e = exp(e - m[dst]); segment sum over dst (atomic).
__global__ void k_edge_expsum(const long long* __restrict__ ei, int E,
                              int Etot, float* __restrict__ eb,
                              const float* __restrict__ mb,
                              float* __restrict__ zb, int H) {
  int idx = blockIdx.x * blockDim.x + threadIdx.x;
  if (idx >= Etot * H) return;
  int e = idx / H, h = idx - e * H;
  int dst = (e < E) ? (int)ei[E + e] : (e - E);
  float w = expf(eb[idx] - mb[dst * H + h]);
  eb[idx] = w;
  atomicAdd(&zb[dst * H + h], w);
}

// Edge pass 3 (big layers, F = H*C = 1024): acc[dst,f] += alpha*xl[src,f].
// One 256-thread block per edge.
__global__ __launch_bounds__(256) void k_edge_aggregate(
    const long long* __restrict__ ei, int E, int Etot,
    const float* __restrict__ xl, const float* __restrict__ eb,
    const float* __restrict__ zb, float* __restrict__ acc, int H, int C) {
  int e = blockIdx.x;
  int t = threadIdx.x;
  int src, dst;
  if (e < E) { src = (int)ei[e]; dst = (int)ei[E + e]; }
  else       { src = dst = e - E; }
  const int F = H * C;
  const float* xr = xl + (size_t)src * F;
  float* ar = acc + (size_t)dst * F;
  __builtin_prefetch(xr + t, 0, 1);                      // global_prefetch_b8
  for (int f = t; f < F; f += 256) {
    int h = f / C;
    float alpha = eb[(size_t)e * H + h] / (zb[dst * H + h] + 1e-16f);
    atomicAdd(&ar[f], alpha * xr[f]);
  }
}

// ---------------------------------------------------------------------------
// conv4 (C==1, H heads): xl4[n,h] = <h_in[n,:], W4[:,h]>, lin4 = <h_in, lw4>.
// One block (8 waves) per node: waves 0..H-1 do heads, wave 6 does skip.
// ---------------------------------------------------------------------------
__global__ __launch_bounds__(256) void k_conv4_node(
    const float* __restrict__ hin, const float* __restrict__ W4,
    const float* __restrict__ lw4, const float* __restrict__ as4,
    const float* __restrict__ ad4, float* __restrict__ xl4,
    float* __restrict__ s4, float* __restrict__ d4,
    float* __restrict__ lin4, int F, int H) {
  int n    = blockIdx.x;
  int wave = threadIdx.x >> 5;
  int lane = threadIdx.x & 31;
  const float* row = hin + (size_t)n * F;
  if (wave < H) {
    float s = 0.0f;
    for (int k = lane; k < F; k += 32) s += row[k] * W4[k * H + wave];
    for (int off = 16; off; off >>= 1) s += __shfl_down(s, off, 32);
    if (lane == 0) {
      xl4[n * H + wave] = s;
      s4[n * H + wave]  = s * as4[wave];
      d4[n * H + wave]  = s * ad4[wave];
    }
  } else if (wave == 6) {
    float s = 0.0f;
    for (int k = lane; k < F; k += 32) s += row[k] * lw4[k];
    for (int off = 16; off; off >>= 1) s += __shfl_down(s, off, 32);
    if (lane == 0) lin4[n] = s;
  }
}

// conv4 scatter: acc4[dst,h] += alpha * xl4[src,h]. One thread per edge.
__global__ void k_edge_aggregate4(const long long* __restrict__ ei, int E,
                                  int Etot, const float* __restrict__ xl4,
                                  const float* __restrict__ eb,
                                  const float* __restrict__ zb,
                                  float* __restrict__ acc4, int H) {
  int e = blockIdx.x * blockDim.x + threadIdx.x;
  if (e >= Etot) return;
  int src, dst;
  if (e < E) { src = (int)ei[e]; dst = (int)ei[E + e]; }
  else       { src = dst = e - E; }
  for (int h = 0; h < H; ++h) {
    float alpha = eb[(size_t)e * H + h] / (zb[dst * H + h] + 1e-16f);
    atomicAdd(&acc4[dst * H + h], alpha * xl4[src * H + h]);
  }
}

// conv4 epilogue: out[n] = mean_h(acc4) + b4 + lin4 + lb4
__global__ void k_final_out(const float* __restrict__ acc4,
                            const float* __restrict__ lin4,
                            const float* __restrict__ b4,
                            const float* __restrict__ lb4,
                            float* __restrict__ out, int Nn, int H) {
  int n = blockIdx.x * blockDim.x + threadIdx.x;
  if (n >= Nn) return;
  float s = 0.0f;
  for (int h = 0; h < H; ++h) s += acc4[n * H + h];
  out[n] = s / (float)H + b4[0] + lin4[n] + lb4[0];
}

// ---------------------------------------------------------------------------
// Host orchestration
// ---------------------------------------------------------------------------
static inline int cdiv(int a, int b) { return (a + b - 1) / b; }

extern "C" void kernel_launch(void* const* d_in, const int* in_sizes, int n_in,
                              void* d_out, int out_size, void* d_ws,
                              size_t ws_size, hipStream_t stream) {
  (void)n_in; (void)out_size; (void)ws_size;
  const int Nn   = in_sizes[0] / 8;      // 20000
  const int E    = in_sizes[1] / 2;      // 100000
  const int Etot = E + Nn;               // with self-loops
  const int F    = 1024;                 // 4*HC
  const int H    = 4, C = 256, H4 = 6;

  const float*     x  = (const float*)d_in[0];
  const long long* ei = (const long long*)d_in[1];   // int64 [2,E]
  const float *W1 = (const float*)d_in[2],  *as1 = (const float*)d_in[3],
              *ad1 = (const float*)d_in[4], *b1 = (const float*)d_in[5],
              *lw1 = (const float*)d_in[6], *lb1 = (const float*)d_in[7];
  const float *W2 = (const float*)d_in[8],  *as2 = (const float*)d_in[9],
              *ad2 = (const float*)d_in[10], *b2 = (const float*)d_in[11],
              *lw2 = (const float*)d_in[12], *lb2 = (const float*)d_in[13];
  const float *W3 = (const float*)d_in[14], *as3 = (const float*)d_in[15],
              *ad3 = (const float*)d_in[16], *b3 = (const float*)d_in[17],
              *lw3 = (const float*)d_in[18], *lb3 = (const float*)d_in[19];
  const float *W4 = (const float*)d_in[20], *as4 = (const float*)d_in[21],
              *ad4 = (const float*)d_in[22], *b4 = (const float*)d_in[23],
              *lw4 = (const float*)d_in[24], *lb4 = (const float*)d_in[25];
  float* out = (float*)d_out;

  // -------- workspace carve-up (256B aligned).  hbf MUST be first: the
  // async GEMM path may read up to one M-tile (64 rows) past its end, which
  // then lands in the buffers carved below (valid memory, results unused).
  char* wptr = (char*)d_ws;
  auto carve = [&](size_t bytes) {
    void* p = wptr;
    wptr += (bytes + 255) & ~(size_t)255;
    return p;
  };
  unsigned short* hbf  = (unsigned short*)carve((size_t)Nn * F * 2);
  unsigned short* wbf  = (unsigned short*)carve((size_t)F * F * 2);  // [N,K]
  unsigned short* lwbf = (unsigned short*)carve((size_t)F * F * 2);  // [N,K]
  float* xl   = (float*)carve((size_t)Nn * F * 4);
  float* skip = (float*)carve((size_t)Nn * F * 4);
  float* acc  = (float*)carve((size_t)Nn * F * 4);
  float* hA   = (float*)carve((size_t)Nn * F * 4);
  float* hB   = (float*)carve((size_t)Nn * F * 4);
  float* sn   = (float*)carve((size_t)Nn * 8 * 4);
  float* dn   = (float*)carve((size_t)Nn * 8 * 4);
  float* mb   = (float*)carve((size_t)Nn * 8 * 4);
  float* zb   = (float*)carve((size_t)Nn * 8 * 4);
  float* eb   = (float*)carve((size_t)Etot * 8 * 4);

  const float NEG_INF = -__builtin_inff();

  // shared attention pipeline (segment softmax over dst) for xl with (h,c)
  auto run_attention = [&](const float* xlv, const float* a_s,
                           const float* a_d, int hh, int cc) {
    k_node_attn<<<cdiv(Nn * hh, 8), 256, 0, stream>>>(xlv, a_s, a_d, sn, dn,
                                                      Nn, hh, cc);
    k_fill_f32<<<cdiv(Nn * hh, 256), 256, 0, stream>>>(mb, NEG_INF, Nn * hh);
    k_fill_f32<<<cdiv(Nn * hh, 256), 256, 0, stream>>>(zb, 0.0f, Nn * hh);
    k_edge_score_max<<<cdiv(Etot * hh, 256), 256, 0, stream>>>(
        ei, E, Etot, sn, dn, eb, mb, hh);
    k_edge_expsum<<<cdiv(Etot * hh, 256), 256, 0, stream>>>(ei, E, Etot, eb,
                                                            mb, zb, hh);
    k_fill_f32<<<cdiv(Nn * F, 256), 256, 0, stream>>>(acc, 0.0f, Nn * F);
    k_edge_aggregate<<<Etot, 256, 0, stream>>>(ei, E, Etot, xlv, eb, zb, acc,
                                               hh, cc);
  };

  // big GAT layer (F -> F, H=4 concat) with WMMA GEMMs
  auto run_big_layer = [&](const float* hin, float* hout, const float* W,
                           const float* a_s, const float* a_d, const float* b,
                           const float* lw, const float* lb) {
    k_f32_to_bf16<<<cdiv(Nn * F, 256), 256, 0, stream>>>(hin, hbf, Nn * F);
    dim3 gt(F / 32, F / 32);
    k_f32_to_bf16_t<<<gt, 256, 0, stream>>>(W,  wbf,  F, F);   // -> [N,K]
    k_f32_to_bf16_t<<<gt, 256, 0, stream>>>(lw, lwbf, F, F);   // -> [N,K]
    dim3 g(cdiv(Nn, GEMM_BM), F / GEMM_BN);
    k_gemm_bf16_wmma<<<g, 256, 0, stream>>>(hbf, wbf,  xl,   Nn, F, F);
    k_gemm_bf16_wmma<<<g, 256, 0, stream>>>(hbf, lwbf, skip, Nn, F, F);
    run_attention(xl, a_s, a_d, H, C);
    k_epilogue_elu<<<cdiv(Nn * F, 256), 256, 0, stream>>>(acc, skip, b, lb,
                                                          hout, Nn * F, F);
  };

  // ---- layer 1: GAT(8 -> 1024 concat), plain small-K GEMMs ----
  k_gemm_smallk<<<cdiv(Nn * F, 256), 256, 0, stream>>>(x, W1,  xl,   Nn, F, 8);
  k_gemm_smallk<<<cdiv(Nn * F, 256), 256, 0, stream>>>(x, lw1, skip, Nn, F, 8);
  run_attention(xl, as1, ad1, H, C);
  k_epilogue_elu<<<cdiv(Nn * F, 256), 256, 0, stream>>>(acc, skip, b1, lb1,
                                                        hA, Nn * F, F);

  // ---- layers 2 & 3: WMMA path ----
  run_big_layer(hA, hB, W2, as2, ad2, b2, lw2, lb2);
  run_big_layer(hB, hA, W3, as3, ad3, b3, lw3, lb3);

  // ---- layer 4: GAT(1024 -> 1, heads=6, mean) + linear skip ----
  float* xl4  = xl;     // [Nn, 6]
  float* lin4 = skip;   // [Nn]
  float* acc4 = acc;    // [Nn, 6]
  k_conv4_node<<<Nn, 256, 0, stream>>>(hA, W4, lw4, as4, ad4, xl4, sn, dn,
                                       lin4, F, H4);
  k_fill_f32<<<cdiv(Nn * H4, 256), 256, 0, stream>>>(mb, NEG_INF, Nn * H4);
  k_fill_f32<<<cdiv(Nn * H4, 256), 256, 0, stream>>>(zb, 0.0f, Nn * H4);
  k_edge_score_max<<<cdiv(Etot * H4, 256), 256, 0, stream>>>(ei, E, Etot, sn,
                                                             dn, eb, mb, H4);
  k_edge_expsum<<<cdiv(Etot * H4, 256), 256, 0, stream>>>(ei, E, Etot, eb, mb,
                                                          zb, H4);
  k_fill_f32<<<cdiv(Nn * H4, 256), 256, 0, stream>>>(acc4, 0.0f, Nn * H4);
  k_edge_aggregate4<<<cdiv(Etot, 256), 256, 0, stream>>>(ei, E, Etot, xl4, eb,
                                                         zb, acc4, H4);
  k_final_out<<<cdiv(Nn, 256), 256, 0, stream>>>(acc4, lin4, b4, lb4, out, Nn,
                                                 H4);
}
